// ProteinGenerator_77859167142547
// MI455X (gfx1250) — matrix operand order repaired and checked by reading, not compiled, not run on hardware
//
#include <hip/hip_runtime.h>
#include <hip/hip_bf16.h>
#include <stdint.h>

// ---------------------------------------------------------------------------
// Model dims
// ---------------------------------------------------------------------------
#define BATCH   64
#define SEQ     512
#define HID     256
#define G4      1024          // 4*HID
#define EMB     16
#define VOCAB   23
#define LAYERS  8
#define MROWS   (BATCH*SEQ)   // 32768 GEMM rows
#define HSTRIDE 264           // padded LDS row stride (bf16) -> banks spread 4m mod 64

typedef __bf16 bf16;
typedef __attribute__((ext_vector_type(16))) __bf16 v16bf;
typedef __attribute__((ext_vector_type(8)))  __bf16 v8bf;
typedef __attribute__((ext_vector_type(8)))  float  v8f;

// ---------------------------------------------------------------------------
// helpers
// ---------------------------------------------------------------------------
__device__ inline v16bf load_frag(const bf16* p0, const bf16* p1) {
  v8bf a = *(const v8bf*)p0;
  v8bf b = *(const v8bf*)p1;
  return __builtin_shufflevector(a, b, 0,1,2,3,4,5,6,7,8,9,10,11,12,13,14,15);
}

__device__ inline v8f wmma_bf16(v16bf a, v16bf b, v8f c) {
  // D = A(16x32 bf16) x B(32x16 bf16) + C(16x16 f32)
  return __builtin_amdgcn_wmma_f32_16x16x32_bf16(false, a, false, b, (short)0, c,
                                                 false, false);
}

__device__ inline float sigm(float x)  { return 1.0f / (1.0f + __expf(-x)); }
__device__ inline float tanh_f(float x){
  x = fminf(fmaxf(x, -15.0f), 15.0f);
  float e = __expf(2.0f * x);
  return (e - 1.0f) / (e + 1.0f);
}

// ---------------------------------------------------------------------------
// Pack an f32 weight matrix W[Nout, Kin] (row-major) into bf16 WMMA-B tiles.
// B-operand layout per 32Kx16N tile: lane = (k_half)*16 + n%16, element j ->
// k = ktile*32 + (lane/16)*16 + j.  16 contiguous bf16 per lane => two b128
// loads at consume time.  dst idx i = ((ktile*Ntiles + ntile)*32 + lane)*16 + j
// ---------------------------------------------------------------------------
__global__ void pack_b_kernel(const float* __restrict__ W, bf16* __restrict__ dst,
                              int Kin, int Ntiles, int Nout, int total) {
  int i = blockIdx.x * 256 + threadIdx.x;
  if (i >= total) return;
  int j    = i & 15;
  int lane = (i >> 4) & 31;
  int tile = i >> 9;
  int nt   = tile % Ntiles;
  int kt   = tile / Ntiles;
  int n    = nt * 16 + (lane & 15);
  int k    = kt * 32 + (lane >> 4) * 16 + j;
  float v  = (k < Kin && n < Nout) ? W[(size_t)n * Kin + k] : 0.0f;
  dst[i] = (bf16)v;
}

// ---------------------------------------------------------------------------
// Embedding lookup -> bf16 activations, K padded 16 -> 32 with zeros.
// act0 is [MROWS, 32] row-major, row index = b*SEQ + t.
// ---------------------------------------------------------------------------
__global__ void embed_kernel(const int* __restrict__ x, const float* __restrict__ emb,
                             bf16* __restrict__ act0, int total) {
  int i = blockIdx.x * 256 + threadIdx.x;
  if (i >= total) return;
  int m = i >> 5, col = i & 31;
  float v = (col < EMB) ? emb[x[m] * EMB + col] : 0.0f;
  act0[i] = (bf16)v;
}

// ---------------------------------------------------------------------------
// Input projection GEMM: xg[MROWS, 1024] = act @ Wih^T + (b_ih + b_hh)
// One wave per (16-row Mtile, 4 consecutive Ntiles).  32768 waves; occupancy
// hides the L2 latency here; loads batched before WMMAs per K-slice.
// ---------------------------------------------------------------------------
__global__ __launch_bounds__(256)
void xproj_kernel(const bf16* __restrict__ act, int astride,
                  const bf16* __restrict__ wp, int ktiles,
                  const float* __restrict__ bih, const float* __restrict__ bhh,
                  float* __restrict__ xg) {
  int wid  = blockIdx.x * 8 + (threadIdx.x >> 5);  // 0..32767
  int lane = threadIdx.x & 31;
  int ln   = lane & 15, lh = lane >> 4;
  int mt   = wid >> 4;                             // 0..2047
  int ng   = wid & 15;                             // group of 4 ntiles

  v8f acc[4];
  #pragma unroll
  for (int j = 0; j < 4; ++j) {
    int n = (ng * 4 + j) * 16 + ln;
    float bv = bih[n] + bhh[n];
    #pragma unroll
    for (int r = 0; r < 8; ++r) acc[j][r] = bv;
  }

  const bf16* arow  = act + (size_t)(mt * 16 + ln) * astride + lh * 16;
  const bf16* bbase = wp + ((size_t)ng * 4 * 32 + lane) * 16;   // + kt*64tiles later
  for (int kt = 0; kt < ktiles; ++kt) {
    const bf16* ap = arow + kt * 32;
    v16bf a = load_frag(ap, ap + 8);
    v16bf bfr[4];
    #pragma unroll
    for (int j = 0; j < 4; ++j) {
      const bf16* bp = bbase + ((size_t)kt * 64 + j) * 512;
      bfr[j] = load_frag(bp, bp + 8);
    }
    #pragma unroll
    for (int j = 0; j < 4; ++j) acc[j] = wmma_bf16(a, bfr[j], acc[j]);
  }

  #pragma unroll
  for (int j = 0; j < 4; ++j) {
    int n = (ng * 4 + j) * 16 + ln;
    #pragma unroll
    for (int r = 0; r < 8; ++r) {
      int m = mt * 16 + lh * 8 + r;
      xg[(size_t)m * G4 + n] = acc[j][r];
    }
  }
}

// ---------------------------------------------------------------------------
// Sequential LSTM scan for one layer.  ONE workgroup, 32 waves on one WGP.
// h lives in LDS (bf16, A-operand row-major, DOUBLE-buffered), c in VGPRs.
// Wave w: Mtile mt = w%4, hidden block hb = w/4 (32 hidden cols) -> owns all
// four gate tiles (i,f,g,o) x 2 subtiles for those columns => gate math is
// fully register-local.
//
// Critical-path discipline per timestep:
//   1. two global_prefetch lines/lane warm this step's 64 xg gate values in
//      the WGP cache while the GEMM runs (they span exactly 64 cachelines
//      per wave = 2 per lane);
//   2. fully unrolled K loop, all 8 B loads clause-issued before the WMMAs
//      so slice k+1 loads overlap slice k WMMAs (partial loadcnt waits);
//   3. gate math (v_exp etc.) runs before the barrier, overlapping slower
//      waves' GEMM tails;
//   4. double-buffered h => ONE barrier per timestep: writes at t target the
//      buffer nobody reads at t; barrier orders write(t)->read(t+1) and
//      read(t)->write(t+1).
// ---------------------------------------------------------------------------
__global__ __launch_bounds__(1024)
void lstm_scan_kernel(const float* __restrict__ xg,
                      const bf16* __restrict__ whhp,
                      bf16* __restrict__ act_out) {
  __shared__ bf16 hbuf[2][BATCH * HSTRIDE];

  int tid  = threadIdx.x;
  int lane = tid & 31;
  int w    = tid >> 5;       // 0..31
  int mt   = w & 3;          // batch tile
  int hb   = w >> 2;         // hidden block (32 cols)
  int ln   = lane & 15, lh = lane >> 4;

  for (int i = tid; i < 2 * BATCH * HSTRIDE; i += 1024)
    ((bf16*)hbuf)[i] = (bf16)0.0f;
  __syncthreads();

  float c[2][8];
  #pragma unroll
  for (int s = 0; s < 2; ++s)
    #pragma unroll
    for (int r = 0; r < 8; ++r) c[s][r] = 0.0f;

  const int   arow  = mt * 16 + ln;                  // A-operand row, this lane
  const bf16* bwave = whhp + ((size_t)hb * 2 * 32 + lane) * 16; // nt = hb*2 base

  for (int t = 0; t < SEQ; ++t) {
    const bf16* rb = hbuf[t & 1];            // read h(t-1)
    bf16*       wbuf = hbuf[(t + 1) & 1];    // write h(t)

    // warm this step's gate lines in the WGP cache (2 lines per lane:
    // row mt*16+(lane&15), gates {lane>>4, 2+(lane>>4)}, cols hb*32..+31)
    const float* pf = xg + ((size_t)(mt * 16 + (lane & 15)) * SEQ + t) * G4
                         + (lane >> 4) * HID + hb * 32;
    __builtin_prefetch(pf, 0, 0);
    __builtin_prefetch(pf + 2 * HID, 0, 0);

    v8f acc[4][2];
    #pragma unroll
    for (int g = 0; g < 4; ++g)
      #pragma unroll
      for (int s = 0; s < 2; ++s)
        #pragma unroll
        for (int r = 0; r < 8; ++r) acc[g][s][r] = 0.0f;

    // ---- GEMM: h[64,256] @ Whh^T -> this wave's 8 gate tiles ----
    const bf16* apb = rb + arow * HSTRIDE + lh * 16;
    #pragma unroll
    for (int kb = 0; kb < 8; ++kb) {
      const bf16* ap = apb + kb * 32;
      v16bf a = load_frag(ap, ap + 8);
      v16bf bfr[8];
      const bf16* bb = bwave + (size_t)kb * 64 * 512;   // this K-slice, nt=hb*2
      #pragma unroll
      for (int g = 0; g < 4; ++g)
        #pragma unroll
        for (int s = 0; s < 2; ++s)
          bfr[g * 2 + s] = load_frag(bb + (g * 16 + s) * 512,
                                     bb + (g * 16 + s) * 512 + 8);
      #pragma unroll
      for (int g = 0; g < 4; ++g)
        #pragma unroll
        for (int s = 0; s < 2; ++s)
          acc[g][s] = wmma_bf16(a, bfr[g * 2 + s], acc[g][s]);
    }

    // ---- gates + state update (register/global only; no LDS) ----
    bf16 hv[2][8];
    #pragma unroll
    for (int s = 0; s < 2; ++s) {
      int col = hb * 32 + s * 16 + ln;
      #pragma unroll
      for (int r = 0; r < 8; ++r) {
        int m = mt * 16 + lh * 8 + r;
        const float* xr = xg + ((size_t)m * SEQ + t) * G4;
        float gi = acc[0][s][r] + xr[0 * HID + col];
        float gf = acc[1][s][r] + xr[1 * HID + col];
        float gg = acc[2][s][r] + xr[2 * HID + col];
        float go = acc[3][s][r] + xr[3 * HID + col];
        float ii = sigm(gi), ff = sigm(gf), oo = sigm(go);
        float cc = ff * c[s][r] + ii * tanh_f(gg);
        c[s][r] = cc;
        hv[s][r] = (bf16)(oo * tanh_f(cc));
      }
    }

    // ---- publish h(t) into the write buffer + layer output ----
    #pragma unroll
    for (int s = 0; s < 2; ++s) {
      int col = hb * 32 + s * 16 + ln;
      #pragma unroll
      for (int r = 0; r < 8; ++r) {
        int m = mt * 16 + lh * 8 + r;
        wbuf[m * HSTRIDE + col] = hv[s][r];
        act_out[((size_t)m * SEQ + t) * HID + col] = hv[s][r];
      }
    }

    __syncthreads();   // single barrier: write(t) visible, read(t) finished
  }
}

// ---------------------------------------------------------------------------
// Final FC: out[MROWS, 23] = act @ fc_w^T + fc_b.  One wave per Mtile.
// ---------------------------------------------------------------------------
__global__ __launch_bounds__(256)
void fc_kernel(const bf16* __restrict__ act, const bf16* __restrict__ wfc,
               const float* __restrict__ fcb, float* __restrict__ out) {
  int wid  = blockIdx.x * 8 + (threadIdx.x >> 5);  // 0..2047 (Mtile)
  int lane = threadIdx.x & 31;
  int ln   = lane & 15, lh = lane >> 4;
  int mt   = wid;

  v8f acc[2];
  #pragma unroll
  for (int j = 0; j < 2; ++j) {
    int n = j * 16 + ln;
    float bv = (n < VOCAB) ? fcb[n] : 0.0f;
    #pragma unroll
    for (int r = 0; r < 8; ++r) acc[j][r] = bv;
  }

  const bf16* arow = act + (size_t)(mt * 16 + ln) * HID + lh * 16;
  for (int kt = 0; kt < 8; ++kt) {
    const bf16* ap = arow + kt * 32;
    v16bf a = load_frag(ap, ap + 8);
    v16bf bfr[2];
    #pragma unroll
    for (int j = 0; j < 2; ++j) {
      const bf16* bp = wfc + ((size_t)(kt * 2 + j) * 32 + lane) * 16;
      bfr[j] = load_frag(bp, bp + 8);
    }
    #pragma unroll
    for (int j = 0; j < 2; ++j) acc[j] = wmma_bf16(a, bfr[j], acc[j]);
  }

  #pragma unroll
  for (int j = 0; j < 2; ++j) {
    int n = j * 16 + ln;
    if (n < VOCAB) {
      #pragma unroll
      for (int r = 0; r < 8; ++r) {
        int m = mt * 16 + lh * 8 + r;
        out[(size_t)m * VOCAB + n] = acc[j][r];
      }
    }
  }
}

// ---------------------------------------------------------------------------
// Host orchestration (graph-capture safe: only kernel launches on `stream`)
// ---------------------------------------------------------------------------
extern "C" void kernel_launch(void* const* d_in, const int* in_sizes, int n_in,
                              void* d_out, int out_size, void* d_ws, size_t ws_size,
                              hipStream_t stream) {
  (void)in_sizes; (void)n_in; (void)out_size; (void)ws_size;

  const int*   x    = (const int*)  d_in[0];
  const float* emb  = (const float*)d_in[1];
  const float* Wih0 = (const float*)d_in[2];
  const float* WihR = (const float*)d_in[3];
  const float* Whh  = (const float*)d_in[4];
  const float* bih  = (const float*)d_in[5];
  const float* bhh  = (const float*)d_in[6];
  const float* fcw  = (const float*)d_in[7];
  const float* fcb  = (const float*)d_in[8];
  float* out = (float*)d_out;

  // workspace carve-up (all 256B aligned)
  char* ws = (char*)d_ws;
  size_t off = 0;
  auto carve = [&](size_t bytes) -> char* {
    char* p = ws + off;
    off += (bytes + 255) & ~(size_t)255;
    return p;
  };
  const size_t WHH_ELEMS = (size_t)8 * 64 * 512;               // 262144 per layer
  bf16* whhp  = (bf16*)carve((size_t)LAYERS * WHH_ELEMS * 2);  // 4 MB
  bf16* wih0p = (bf16*)carve((size_t)1 * 64 * 512 * 2);        // 64 KB
  bf16* wihrp = (bf16*)carve((size_t)(LAYERS-1) * WHH_ELEMS * 2); // 3.5 MB
  bf16* fcp   = (bf16*)carve((size_t)8 * 2 * 512 * 2);         // 16 KB
  bf16* act0  = (bf16*)carve((size_t)MROWS * 32 * 2);          // 2 MB
  bf16* acta  = (bf16*)carve((size_t)MROWS * HID * 2);         // 16 MB
  bf16* actb  = (bf16*)carve((size_t)MROWS * HID * 2);         // 16 MB
  float* xg   = (float*)carve((size_t)MROWS * G4 * 4);         // 128 MB

  // ---- pack all weights to bf16 WMMA-B layout ----
  for (int l = 0; l < LAYERS; ++l)
    pack_b_kernel<<<(int)((WHH_ELEMS + 255) / 256), 256, 0, stream>>>(
        Whh + (size_t)l * G4 * HID, whhp + (size_t)l * WHH_ELEMS,
        HID, 64, G4, (int)WHH_ELEMS);
  pack_b_kernel<<<(32768 + 255) / 256, 256, 0, stream>>>(
      Wih0, wih0p, EMB, 64, G4, 32768);
  for (int l = 0; l < LAYERS - 1; ++l)
    pack_b_kernel<<<(int)((WHH_ELEMS + 255) / 256), 256, 0, stream>>>(
        WihR + (size_t)l * G4 * HID, wihrp + (size_t)l * WHH_ELEMS,
        HID, 64, G4, (int)WHH_ELEMS);
  pack_b_kernel<<<(8192 + 255) / 256, 256, 0, stream>>>(
      fcw, fcp, HID, 2, VOCAB, 8192);

  // ---- embedding ----
  embed_kernel<<<(MROWS * 32) / 256, 256, 0, stream>>>(x, emb, act0, MROWS * 32);

  // ---- layers ----
  bf16* ain  = act0;  int astride = 32;  int ktiles = 1;
  bf16* aout = acta;
  for (int l = 0; l < LAYERS; ++l) {
    const bf16* wp = (l == 0) ? wih0p : (wihrp + (size_t)(l - 1) * WHH_ELEMS);
    xproj_kernel<<<4096, 256, 0, stream>>>(ain, astride, wp, ktiles,
                                           bih + (size_t)l * G4,
                                           bhh + (size_t)l * G4, xg);
    lstm_scan_kernel<<<1, 1024, 0, stream>>>(xg, whhp + (size_t)l * WHH_ELEMS, aout);
    ain = aout; astride = HID; ktiles = 8;
    aout = (aout == acta) ? actb : acta;
  }

  // ---- final FC ----
  fc_kernel<<<256, 256, 0, stream>>>(ain, fcp, fcb, out);
}
